// CondAttLSTM_29429115912193
// MI455X (gfx1250) — matrix operand order repaired and checked by reading, not compiled, not run on hardware
//
#include <hip/hip_runtime.h>

// ---------------- CDNA5 WMMA plumbing (gfx1250, wave32) ----------------
typedef __attribute__((ext_vector_type(16))) __bf16 v16bf;
typedef __attribute__((ext_vector_type(8)))  float  v8f;

__device__ __forceinline__ v8f wmma_bf16(v16bf a, v16bf b, v8f c) {
    // D = A(16x32 bf16) * B(32x16 bf16) + C(16x16 f32)
    return __builtin_amdgcn_wmma_f32_16x16x32_bf16(
        false, a, false, b, (short)0, c, false, false);
}

// A tile (16x32, bf16) from row-major LDS [16][stride].
// lanes 0-15 => M=lane, VGPR j holds K=16*half+2*j(+1), half = lane>>4.
__device__ __forceinline__ v16bf load_a_tile(const __bf16* base, int stride,
                                             int k0, int lane) {
    const int r = lane & 15, hl = lane >> 4;
    v16bf a;
#pragma unroll
    for (int j = 0; j < 8; ++j) {
        const int bk = k0 + 16 * hl + 2 * j;
        a[2 * j]     = base[r * stride + bk];
        a[2 * j + 1] = base[r * stride + bk + 1];
    }
    return a;
}

// B tile (32x16, bf16): element (k, n) = W[n0+n][k0+k], W row-major [N][K].
// Lanes 0-15 hold K=0..15 (VGPR j => K=2j,2j+1), lanes 16-31 hold K=16..31.
__device__ __forceinline__ v16bf load_b_tile(const __bf16* W, int n0, int k0,
                                             int K, int lane) {
    const int n = lane & 15, hl = lane >> 4;
    v16bf b;
    const __bf16* row = W + (size_t)(n0 + n) * K + k0 + 16 * hl;
#pragma unroll
    for (int j = 0; j < 8; ++j) {
        b[2 * j]     = row[2 * j];
        b[2 * j + 1] = row[2 * j + 1];
    }
    return b;
}

__device__ __forceinline__ float sigf(float x) { return 1.0f / (1.0f + expf(-x)); }

// ---------------- problem constants ----------------
#define PB   32
#define PT   128
#define PDIN 512
#define PD   512
#define PDC  512
#define PL   64
#define PA   128
#define PG4  2048   // 4*D

// ---------------- prep kernels ----------------
__global__ void cvt_f32_bf16(__bf16* __restrict__ dst, const float* __restrict__ src, int n) {
    for (int i = blockIdx.x * blockDim.x + threadIdx.x; i < n; i += gridDim.x * blockDim.x)
        dst[i] = (__bf16)src[i];
}

__global__ void init_histproj(float* __restrict__ hp, const float* __restrict__ bhh, int n) {
    for (int i = blockIdx.x * blockDim.x + threadIdx.x; i < n; i += gridDim.x * blockDim.x)
        hp[i] = bhh[i & (PA - 1)];
}

// ---------------- generic 16-row bf16 WMMA GEMM: out = A(f32)*W^T(bf16) + bias ----------
// block = 256 threads (8 waves). blockIdx.x -> M/16 tile, blockIdx.y -> 128-col group.
__global__ void gemm16_bf16(float* __restrict__ out, const float* __restrict__ Asrc,
                            const __bf16* __restrict__ Wb, const float* __restrict__ bias,
                            int M, int N, int K) {
    __shared__ __bf16 shA[16 * 512];
    const int tid  = threadIdx.x;
    const int wave = tid >> 5;
    const int lane = tid & 31;
    const int m0   = blockIdx.x * 16;

    for (int idx = tid; idx < 16 * K; idx += 256) {
        const int r = idx / K, k = idx - r * K;
        shA[r * K + k] = (__bf16)Asrc[(size_t)(m0 + r) * K + k];
    }
    __syncthreads();

    const int n0 = blockIdx.y * 128 + wave * 16;
    if (n0 < N) {
        v8f acc = {0.f, 0.f, 0.f, 0.f, 0.f, 0.f, 0.f, 0.f};
        const int kt = K >> 5;
        for (int kk = 0; kk < kt; ++kk) {
            __builtin_prefetch(Wb + (size_t)(n0 + (lane & 15)) * K + ((kk + 1) & (kt - 1)) * 32, 0, 3);
            v16bf a = load_a_tile(shA, K, kk * 32, lane);
            v16bf b = load_b_tile(Wb, n0, kk * 32, K, lane);
            acc = wmma_bf16(a, b, acc);
        }
        const int n = lane & 15, hl = lane >> 4;
        const float bv = bias ? bias[n0 + n] : 0.f;
#pragma unroll
        for (int e = 0; e < 8; ++e) {
            const int m = e + 8 * hl;
            out[(size_t)(m0 + m) * N + n0 + n] = acc[e] + bv;
        }
    }
}

// ---------------- the sequential scan (one block per 16 batch rows) ----------------
__global__ void __launch_bounds__(512)
cond_att_lstm_scan(const float* __restrict__ context,   // [B,L,DC]
                   const float* __restrict__ h0,        // [B,D]
                   const float* __restrict__ wa,        // [A]
                   const float* __restrict__ ba1,       // [1]
                   const float* __restrict__ bhh,       // [A]
                   const float* __restrict__ wha,       // [A]
                   const float* __restrict__ bha1,      // [1]
                   const int*   __restrict__ parent_t,  // [B,T]
                   const float* __restrict__ xproj,     // [B,T,4D] ws (incl bx)
                   const float* __restrict__ ctxatt,    // [B,L,A]  ws (incl bac)
                   float*       __restrict__ histproj,  // [B,T,A]  ws (init bhh)
                   const __bf16* __restrict__ Uhb, const __bf16* __restrict__ Ccb,
                   const __bf16* __restrict__ Phb, const __bf16* __restrict__ Hhb,
                   const __bf16* __restrict__ Wahb, const __bf16* __restrict__ Whqb,
                   const __bf16* __restrict__ Whhb,
                   float* __restrict__ out_h,           // [B,T,D]
                   float* __restrict__ out_ctx) {       // [B,T,DC]
    __shared__ __bf16 sh_h[16 * PD];       // 16 KB  current hidden (WMMA A operand)
    __shared__ __bf16 sh_stage[16 * PD];   // 16 KB  ctx / par / hctx staging (A operand)
    __shared__ float  sh_q[16 * PA];       //  8 KB  h @ Wah^T
    __shared__ float  sh_q2[16 * PA];      //  8 KB  h @ Whq^T
    __shared__ float  sh_sc[16 * PA];      //  8 KB  scores / softmax weights

    const int tid   = threadIdx.x;
    const int wave  = tid >> 5;
    const int lane  = tid & 31;
    const int n     = lane & 15;
    const int hl    = lane >> 4;
    const int bbase = blockIdx.x * 16;
    const float ba  = ba1[0];
    const float bha = bha1[0];

    // persistent cell state in registers: wave owns columns [wave*32, wave*32+32)
    v8f creg0 = {0.f,0.f,0.f,0.f,0.f,0.f,0.f,0.f};
    v8f creg1 = {0.f,0.f,0.f,0.f,0.f,0.f,0.f,0.f};

    // stage h0 as bf16
    for (int idx = tid; idx < 16 * PD; idx += 512) {
        const int b = idx >> 9, d = idx & (PD - 1);
        sh_h[idx] = (__bf16)h0[(size_t)(bbase + b) * PD + d];
    }
    __syncthreads();

    for (int t = 0; t < PT; ++t) {
        // ---- P1: gate accumulators = h @ Uh^T (tiles: j = g*2+p, cols g*512+wave*32+p*16)
        v8f acc[8];
#pragma unroll
        for (int j = 0; j < 8; ++j) acc[j] = (v8f){0.f,0.f,0.f,0.f,0.f,0.f,0.f,0.f};
        for (int kk = 0; kk < 16; ++kk) {
            v16bf a = load_a_tile(sh_h, PD, kk * 32, lane);
#pragma unroll
            for (int j = 0; j < 8; ++j) {
                const int nc0 = (j >> 1) * 512 + wave * 32 + (j & 1) * 16;
                v16bf bt = load_b_tile(Uhb, nc0, kk * 32, PD, lane);
                acc[j] = wmma_bf16(a, bt, acc[j]);
            }
        }

        // ---- P2: q = h@Wah^T (waves 0..7), q2 = h@Whq^T (waves 8..15)
        {
            const __bf16* Wq = (wave < 8) ? Wahb : Whqb;
            float* dstq      = (wave < 8) ? sh_q : sh_q2;
            const int qn0    = (wave & 7) * 16;
            v8f qa = {0.f,0.f,0.f,0.f,0.f,0.f,0.f,0.f};
            for (int kk = 0; kk < 16; ++kk) {
                v16bf a = load_a_tile(sh_h, PD, kk * 32, lane);
                v16bf bt = load_b_tile(Wq, qn0, kk * 32, PD, lane);
                qa = wmma_bf16(a, bt, qa);
            }
#pragma unroll
            for (int e = 0; e < 8; ++e) dstq[(e + 8 * hl) * PA + qn0 + n] = qa[e];
        }
        __syncthreads();

        // ---- P3: context attention (per-batch-row matvecs -> VALU)
        for (int idx = tid; idx < 16 * PL; idx += 512) {
            const int b = idx >> 6, l = idx & (PL - 1);
            const float* ca = ctxatt + ((size_t)(bbase + b) * PL + l) * PA;
            float s = 0.f;
            for (int aix = 0; aix < PA; ++aix)
                s += tanhf(ca[aix] + sh_q[b * PA + aix]) * wa[aix];
            sh_sc[b * PA + l] = s + ba;
        }
        __syncthreads();
        if (tid < 16) {
            float m = -1e30f;
            for (int l = 0; l < PL; ++l) m = fmaxf(m, sh_sc[tid * PA + l]);
            float sum = 0.f;
            for (int l = 0; l < PL; ++l) sum += expf(sh_sc[tid * PA + l] - m);
            const float inv = 1.f / sum;
            for (int l = 0; l < PL; ++l)
                sh_sc[tid * PA + l] = expf(sh_sc[tid * PA + l] - m) * inv;
        }
        __syncthreads();
        for (int idx = tid; idx < 16 * PDC; idx += 512) {
            const int b = idx >> 9, d = idx & (PDC - 1);
            const float* crow = context + (size_t)(bbase + b) * PL * PDC + d;
            float v = 0.f;
            for (int l = 0; l < PL; ++l) v += sh_sc[b * PA + l] * crow[(size_t)l * PDC];
            out_ctx[((size_t)(bbase + b) * PT + t) * PDC + d] = v;
            sh_stage[idx] = (__bf16)v;
        }
        __syncthreads();

        // ---- P4: gates += ctx_vec @ Cc^T
        for (int kk = 0; kk < 16; ++kk) {
            v16bf a = load_a_tile(sh_stage, PD, kk * 32, lane);
#pragma unroll
            for (int j = 0; j < 8; ++j) {
                const int nc0 = (j >> 1) * 512 + wave * 32 + (j & 1) * 16;
                v16bf bt = load_b_tile(Ccb, nc0, kk * 32, PD, lane);
                acc[j] = wmma_bf16(a, bt, acc[j]);
            }
        }
        __syncthreads();

        // ---- P5: parent hidden gather (irregular)
        for (int idx = tid; idx < 16 * PD; idx += 512) {
            const int b = idx >> 9, d = idx & (PD - 1);
            const int pt = parent_t[(size_t)(bbase + b) * PT + t];
            float v = 0.f;
            if (t > 0 && pt < t)
                v = out_h[((size_t)(bbase + b) * PT + pt) * PD + d];
            sh_stage[idx] = (__bf16)v;
        }
        __syncthreads();

        // ---- P6: gates += par_h @ Ph^T
        for (int kk = 0; kk < 16; ++kk) {
            v16bf a = load_a_tile(sh_stage, PD, kk * 32, lane);
#pragma unroll
            for (int j = 0; j < 8; ++j) {
                const int nc0 = (j >> 1) * 512 + wave * 32 + (j & 1) * 16;
                v16bf bt = load_b_tile(Phb, nc0, kk * 32, PD, lane);
                acc[j] = wmma_bf16(a, bt, acc[j]);
            }
        }
        __syncthreads();

        // ---- P7: history attention (cached projections + incremental softmax)
        for (int idx = tid; idx < 16 * PT; idx += 512) {
            const int b = idx >> 7, s = idx & (PT - 1);
            const float* hp = histproj + ((size_t)(bbase + b) * PT + s) * PA;
            float sc = 0.f;
            for (int aix = 0; aix < PA; ++aix)
                sc += tanhf(hp[aix] + sh_q2[b * PA + aix]) * wha[aix];
            sh_sc[idx] = sc + bha;
        }
        __syncthreads();
        if (tid < 16) {
            float m = -1e30f;
            for (int s = 0; s < PT; ++s) m = fmaxf(m, sh_sc[tid * PA + s]);
            float sum = 0.f;
            for (int s = 0; s < PT; ++s) sum += expf(sh_sc[tid * PA + s] - m);
            const float inv = 1.f / (sum + 1e-7f);
            for (int s = 0; s < PT; ++s)
                sh_sc[tid * PA + s] = expf(sh_sc[tid * PA + s] - m) * inv;
        }
        __syncthreads();
        for (int idx = tid; idx < 16 * PD; idx += 512) {
            const int b = idx >> 9, d = idx & (PD - 1);
            float v = 0.f;
            if (t > 0) {
                const float* hrow = out_h + (size_t)(bbase + b) * PT * PD + d;
                for (int s = 0; s < t; ++s)
                    v += sh_sc[b * PA + s] * hrow[(size_t)s * PD];
            }
            sh_stage[idx] = (__bf16)v;
        }
        __syncthreads();

        // ---- P8: gates += h_ctx @ Hh^T
        for (int kk = 0; kk < 16; ++kk) {
            v16bf a = load_a_tile(sh_stage, PD, kk * 32, lane);
#pragma unroll
            for (int j = 0; j < 8; ++j) {
                const int nc0 = (j >> 1) * 512 + wave * 32 + (j & 1) * 16;
                v16bf bt = load_b_tile(Hhb, nc0, kk * 32, PD, lane);
                acc[j] = wmma_bf16(a, bt, acc[j]);
            }
        }

        // ---- P9: + x_proj, LSTM nonlinearity (register-resident, wave-local)
#pragma unroll
        for (int j = 0; j < 8; ++j) {
            const int colg = (j >> 1) * 512 + wave * 32 + (j & 1) * 16 + n;
#pragma unroll
            for (int e = 0; e < 8; ++e) {
                const int m = e + 8 * hl;
                acc[j][e] += xproj[((size_t)(bbase + m) * PT + t) * PG4 + colg];
            }
        }
#pragma unroll
        for (int p = 0; p < 2; ++p) {
            const int col = wave * 32 + p * 16 + n;
#pragma unroll
            for (int e = 0; e < 8; ++e) {
                const int m = e + 8 * hl;
                const float gi = acc[0 + p][e];
                const float gf = acc[2 + p][e];
                const float gc = acc[4 + p][e];
                const float go = acc[6 + p][e];
                const float cold = (p == 0) ? creg0[e] : creg1[e];
                const float cn = sigf(gf) * cold + sigf(gi) * tanhf(gc);
                const float hn = sigf(go) * tanhf(cn);
                if (p == 0) creg0[e] = cn; else creg1[e] = cn;
                out_h[((size_t)(bbase + m) * PT + t) * PD + col] = hn;
                sh_h[m * PD + col] = (__bf16)hn;
            }
        }
        __syncthreads();

        // ---- P10: cache histproj[t] = h_new @ Whh^T + bhh (waves 0..7)
        if (wave < 8) {
            const int qn0 = wave * 16;
            v8f hp = {0.f,0.f,0.f,0.f,0.f,0.f,0.f,0.f};
            for (int kk = 0; kk < 16; ++kk) {
                v16bf a = load_a_tile(sh_h, PD, kk * 32, lane);
                v16bf bt = load_b_tile(Whhb, qn0, kk * 32, PD, lane);
                hp = wmma_bf16(a, bt, hp);
            }
            const float bv = bhh[qn0 + n];
#pragma unroll
            for (int e = 0; e < 8; ++e) {
                const int m = e + 8 * hl;
                histproj[((size_t)(bbase + m) * PT + t) * PA + qn0 + n] = hp[e] + bv;
            }
        }
        __threadfence();
        __syncthreads();
    }
}

// ---------------- host-side launch ----------------
extern "C" void kernel_launch(void* const* d_in, const int* in_sizes, int n_in,
                              void* d_out, int out_size, void* d_ws, size_t ws_size,
                              hipStream_t stream) {
    (void)in_sizes; (void)n_in; (void)out_size; (void)ws_size;
    const float* X       = (const float*)d_in[0];
    const float* context = (const float*)d_in[1];
    const float* h0      = (const float*)d_in[2];
    const float* Wx      = (const float*)d_in[3];
    const float* bx      = (const float*)d_in[4];
    const float* Uh      = (const float*)d_in[5];
    const float* Cc      = (const float*)d_in[6];
    const float* Ph      = (const float*)d_in[7];
    const float* Hh      = (const float*)d_in[8];
    const float* Wac     = (const float*)d_in[9];
    const float* bac     = (const float*)d_in[10];
    const float* Wah     = (const float*)d_in[11];
    const float* wa      = (const float*)d_in[12];
    const float* ba      = (const float*)d_in[13];
    const float* Whh     = (const float*)d_in[14];
    const float* bhh     = (const float*)d_in[15];
    const float* Whq     = (const float*)d_in[16];
    const float* wha     = (const float*)d_in[17];
    const float* bha     = (const float*)d_in[18];
    const int*   parent  = (const int*)d_in[19];

    float* out_h   = (float*)d_out;                          // [B,T,D]
    float* out_ctx = (float*)d_out + (size_t)PB * PT * PD;   // [B,T,DC]

    // workspace carve-up
    char* ws = (char*)d_ws;
    size_t off = 0;
    auto carve = [&](size_t bytes) { char* p = ws + off; off += (bytes + 255) & ~(size_t)255; return p; };
    __bf16* Wxb  = (__bf16*)carve((size_t)PG4 * PDIN * 2);
    __bf16* Uhb  = (__bf16*)carve((size_t)PG4 * PD * 2);
    __bf16* Ccb  = (__bf16*)carve((size_t)PG4 * PDC * 2);
    __bf16* Phb  = (__bf16*)carve((size_t)PG4 * PD * 2);
    __bf16* Hhb  = (__bf16*)carve((size_t)PG4 * PD * 2);
    __bf16* Wacb = (__bf16*)carve((size_t)PA * PDC * 2);
    __bf16* Wahb = (__bf16*)carve((size_t)PA * PD * 2);
    __bf16* Whhb = (__bf16*)carve((size_t)PA * PD * 2);
    __bf16* Whqb = (__bf16*)carve((size_t)PA * PD * 2);
    float*  xproj    = (float*)carve((size_t)PB * PT * PG4 * 4);
    float*  ctxatt   = (float*)carve((size_t)PB * PL * PA * 4);
    float*  histproj = (float*)carve((size_t)PB * PT * PA * 4);

    // weight conversions f32 -> bf16
    auto cvt = [&](__bf16* dst, const float* src, int nelem) {
        cvt_f32_bf16<<<(nelem + 255) / 256, 256, 0, stream>>>(dst, src, nelem);
    };
    cvt(Wxb,  Wx,  PG4 * PDIN);
    cvt(Uhb,  Uh,  PG4 * PD);
    cvt(Ccb,  Cc,  PG4 * PDC);
    cvt(Phb,  Ph,  PG4 * PD);
    cvt(Hhb,  Hh,  PG4 * PD);
    cvt(Wacb, Wac, PA * PDC);
    cvt(Wahb, Wah, PA * PD);
    cvt(Whhb, Whh, PA * PD);
    cvt(Whqb, Whq, PA * PD);

    // histproj init: rows project of zero hist == bhh
    init_histproj<<<(PB * PT * PA + 255) / 256, 256, 0, stream>>>(histproj, bhh, PB * PT * PA);

    // x_proj = X @ Wx^T + bx   :  [4096 x 2048], K=512
    gemm16_bf16<<<dim3(PB * PT / 16, PG4 / 128), 256, 0, stream>>>(
        xproj, X, Wxb, bx, PB * PT, PG4, PDIN);

    // ctx_att_trans = context @ Wac^T + bac : [2048 x 128], K=512
    gemm16_bf16<<<dim3(PB * PL / 16, 1), 256, 0, stream>>>(
        ctxatt, context, Wacb, bac, PB * PL, PA, PDC);

    // sequential scan: 2 blocks x 512 threads (16 batch rows each)
    cond_att_lstm_scan<<<dim3(PB / 16), 512, 0, stream>>>(
        context, h0, wa, ba, bhh, wha, bha, parent,
        xproj, ctxatt, histproj,
        Uhb, Ccb, Phb, Hhb, Wahb, Whqb, Whhb,
        out_h, out_ctx);
}
